// TestLSTMKeyed_30537217474999
// MI455X (gfx1250) — matrix operand order repaired and checked by reading, not compiled, not run on hardware
//
#include <hip/hip_runtime.h>

typedef __bf16 bf16_t;
typedef __attribute__((ext_vector_type(16))) __bf16 bf16x16;
typedef __attribute__((ext_vector_type(8)))  __bf16 bf16x8;
typedef __attribute__((ext_vector_type(8)))  float  f32x8;
typedef unsigned int u32x4 __attribute__((ext_vector_type(4)));
typedef int          i32x4 __attribute__((ext_vector_type(4)));
typedef int          i32x8 __attribute__((ext_vector_type(8)));

#define B_SZ 64
#define T_SZ 2048
#define I_SZ 256
#define H_SZ 256
#define G_SZ 1024   /* 4*H */
#define C_SZ 10

__device__ __forceinline__ float sigmoidf_(float v) {
  return 1.0f / (1.0f + __expf(-v));
}

// A-fragment (16x32 bf16) from LDS row-major [16][256]:
// lane = row (lane%16); lane-half picks K chunk {kb..kb+15}
__device__ __forceinline__ bf16x16 make_a_lds(const bf16_t* p) {
  bf16x8 lo = *(const bf16x8*)p;
  bf16x8 hi = *(const bf16x8*)(p + 8);
  bf16x16 a;
#pragma unroll
  for (int i = 0; i < 8; ++i) { a[i] = lo[i]; a[i + 8] = hi[i]; }
  return a;
}

// Same A-fragment but converting f32 global data on the fly
__device__ __forceinline__ bf16x16 make_a_f32(const float* __restrict__ p) {
  bf16x16 a;
#pragma unroll
  for (int i = 0; i < 8; ++i) {
    a[i]     = (bf16_t)p[i];
    a[i + 8] = (bf16_t)p[i + 8];
  }
  return a;
}

// ---------------------------------------------------------------------------
// Tensor Data Mover: async 2-D tile load (global -> LDS).
// tile_d1 rows of tile_d0 elements, row stride d0_stride (data_size units).
// ---------------------------------------------------------------------------
__device__ __forceinline__ void tdm_load_tile_2d(
    uint32_t lds_addr, uint64_t gaddr_bytes, uint32_t tile_d0,
    uint32_t tile_d1, uint64_t d0_stride_elems, uint32_t dsize_code) {
  u32x4 g0;
  g0.x = 1u;                                   // count=1, user descriptor
  g0.y = lds_addr;                             // LDS byte address
  g0.z = (uint32_t)gaddr_bytes;                // global addr [31:0]
  g0.w = (uint32_t)((gaddr_bytes >> 32) & 0x01FFFFFFull) | (2u << 30); // type=2
  i32x8 g1;
  g1[0] = (int)(dsize_code << 16);             // data_size; no multicast/pad
  g1[1] = (int)((tile_d0 & 0xFFFFu) << 16);    // tensor_dim0[15:0]
  g1[2] = (int)(((tile_d0 >> 16) & 0xFFFFu) | ((tile_d1 & 0xFFFFu) << 16));
  g1[3] = (int)(((tile_d1 >> 16) & 0xFFFFu) | ((tile_d0 & 0xFFFFu) << 16)); // tile_dim0
  g1[4] = (int)(tile_d1 & 0xFFFFu);            // tile_dim1; tile_dim2=0
  g1[5] = (int)(d0_stride_elems & 0xFFFFFFFFull);      // dim0_stride lo
  g1[6] = (int)((d0_stride_elems >> 32) & 0xFFFFull);  // dim0_stride hi
  g1[7] = 0;
  i32x4 gz = {0, 0, 0, 0};
#if __clang_major__ >= 23
  i32x8 gz8 = {0, 0, 0, 0, 0, 0, 0, 0};
  __builtin_amdgcn_tensor_load_to_lds(g0, g1, gz, gz, gz8, 0);
#else
  __builtin_amdgcn_tensor_load_to_lds(g0, g1, gz, gz, 0);
#endif
}

// ---------------------------------------------------------------------------
// Pack WcT[g][i] = sum_j key[i][j] * W_ih[g][j]  (bf16 out, [1024][256])
// ---------------------------------------------------------------------------
__global__ __launch_bounds__(256) void pack_wct_kernel(
    const float* __restrict__ key, const float* __restrict__ wih,
    bf16_t* __restrict__ wct) {
  __shared__ float wrow[I_SZ];
  const int g = blockIdx.x;
  const int i = threadIdx.x;
  wrow[i] = wih[(size_t)g * I_SZ + i];
  __syncthreads();
  const float* krow = key + (size_t)i * I_SZ;
  float s = 0.0f;
#pragma unroll 8
  for (int j = 0; j < I_SZ; ++j) s += krow[j] * wrow[j];
  wct[(size_t)g * I_SZ + i] = (bf16_t)s;
}

// Convert W_hh f32 -> bf16 (row-major [1024][256], rows contiguous over K)
__global__ __launch_bounds__(256) void pack_whh_kernel(
    const float* __restrict__ whh_f, bf16_t* __restrict__ whh_b) {
  const int idx = blockIdx.x * 256 + threadIdx.x;
  whh_b[idx] = (bf16_t)whh_f[idx];
}

// ---------------------------------------------------------------------------
// x_proj[M=131072][1024] = bf16(x @ WcT^T + b_ih)   (WMMA bf16, f32 accum)
// grid (16, 1024), block 256 = 8 waves; wave -> 16 rows, block -> 64 cols
// ---------------------------------------------------------------------------
__global__ __launch_bounds__(256, 2) void xproj_gemm_kernel(
    const float* __restrict__ x, const bf16_t* __restrict__ wct,
    const float* __restrict__ b_ih, bf16_t* __restrict__ xproj) {
  const int tid = threadIdx.x;
  const int wave = tid >> 5;
  const int lane = tid & 31;
  const int laneHalf = lane >> 4;
  const int lane16 = lane & 15;
  const int rowBase = blockIdx.y * 128 + wave * 16;
  const int nBase = blockIdx.x * 64;

  f32x8 acc[4];
#pragma unroll
  for (int nt = 0; nt < 4; ++nt) acc[nt] = (f32x8){0, 0, 0, 0, 0, 0, 0, 0};

#pragma unroll
  for (int ks = 0; ks < 8; ++ks) {
    const int kb = ks * 32 + laneHalf * 16;
    bf16x16 a = make_a_f32(x + (size_t)(rowBase + lane16) * I_SZ + kb);
#pragma unroll
    for (int nt = 0; nt < 4; ++nt) {
      const int nrow = nBase + nt * 16 + lane16;
      bf16x16 b = *(const bf16x16*)(wct + (size_t)nrow * I_SZ + ks * 32 + laneHalf * 16);
      acc[nt] = __builtin_amdgcn_wmma_f32_16x16x32_bf16(
          false, a, false, b, (short)0, acc[nt], false, false);
    }
  }
#pragma unroll
  for (int nt = 0; nt < 4; ++nt) {
    const int n = nBase + nt * 16 + lane16;
    const float bias = b_ih[n];
#pragma unroll
    for (int v = 0; v < 8; ++v) {
      const int m = rowBase + v + laneHalf * 8;
      xproj[(size_t)m * G_SZ + n] = (bf16_t)(acc[nt][v] + bias);
    }
  }
}

// ---------------------------------------------------------------------------
// Recurrent scan. 4 blocks x 16 batch rows; 512 threads = 16 waves.
// Wave w owns N-tiles {g*16+w} for gates g=0..3 => i/f/g/o for h-columns
// [16w,16w+16) live in the same lane's accumulators; c stays in registers;
// h double-buffered in LDS (bf16). In the PRECOMP variant the per-step gate
// tile x_proj[16x1024] is TDM-prefetched (double-buffered) into LDS.
// ---------------------------------------------------------------------------
template <bool FUSED>
__global__ __launch_bounds__(512, 1) void lstm_scan_kernel(
    const float* __restrict__ x,        // FUSED only
    const bf16_t* __restrict__ xproj,   // !FUSED only
    const bf16_t* __restrict__ wct,     // FUSED only
    const bf16_t* __restrict__ whh,
    const float* __restrict__ b_ih, const float* __restrict__ b_hh,
    float* __restrict__ h_last) {
  __shared__ __align__(128) bf16_t h_sh[2][16 * H_SZ];   // 16 KB
  __shared__ __align__(128) bf16_t xtile[2][16 * G_SZ];  // 64 KB (PRECOMP)

  const int tid = threadIdx.x;
  const int wave = tid >> 5;       // 0..15
  const int lane = tid & 31;
  const int laneHalf = lane >> 4;
  const int lane16 = lane & 15;
  const int batch0 = blockIdx.x * 16;
  const int hg = wave;             // h-column group owned by this wave

  for (int i = tid; i < 16 * H_SZ; i += 512) h_sh[0][i] = (bf16_t)0.0f;

  float bias[4];
#pragma unroll
  for (int g = 0; g < 4; ++g) {
    const int n = g * 256 + hg * 16 + lane16;
    bias[g] = b_hh[n] + (FUSED ? b_ih[n] : 0.0f);
  }

  float cst[8];
#pragma unroll
  for (int v = 0; v < 8; ++v) cst[v] = 0.0f;

  if constexpr (!FUSED) {
    if (wave == 0) {  // TDM prefetch of step-0 gate tile
      tdm_load_tile_2d((uint32_t)(size_t)&xtile[0][0],
                       (uint64_t)(size_t)(xproj + (size_t)batch0 * T_SZ * G_SZ),
                       G_SZ, 16, (uint64_t)T_SZ * G_SZ, /*2-byte*/ 1u);
    }
  }
  __syncthreads();

  for (int t = 0; t < T_SZ; ++t) {
    const bf16_t* hbuf = h_sh[t & 1];
    bf16_t* hnext = h_sh[(t & 1) ^ 1];

    f32x8 acc[4];
#pragma unroll
    for (int g = 0; g < 4; ++g) acc[g] = (f32x8){0, 0, 0, 0, 0, 0, 0, 0};

#pragma unroll
    for (int ks = 0; ks < 8; ++ks) {
      const int kb = ks * 32 + laneHalf * 16;
      bf16x16 a_h = make_a_lds(&hbuf[lane16 * H_SZ + kb]);
      bf16x16 a_x;
      if constexpr (FUSED) {
        a_x = make_a_f32(x + ((size_t)(batch0 + lane16) * T_SZ + t) * I_SZ + kb);
      }
#pragma unroll
      for (int g = 0; g < 4; ++g) {
        const int nrow = (g * 16 + hg) * 16 + lane16;
        bf16x16 b_h = *(const bf16x16*)(whh + (size_t)nrow * H_SZ + ks * 32 + laneHalf * 16);
        acc[g] = __builtin_amdgcn_wmma_f32_16x16x32_bf16(
            false, a_h, false, b_h, (short)0, acc[g], false, false);
        if constexpr (FUSED) {
          bf16x16 b_x = *(const bf16x16*)(wct + (size_t)nrow * I_SZ + ks * 32 + laneHalf * 16);
          acc[g] = __builtin_amdgcn_wmma_f32_16x16x32_bf16(
              false, a_x, false, b_x, (short)0, acc[g], false, false);
        }
      }
    }

    if constexpr (!FUSED) {
      // Prefetch step t+1 tile, then make sure step t's tile has landed.
      if (wave == 0) {
        if (t + 1 < T_SZ) {
          tdm_load_tile_2d(
              (uint32_t)(size_t)&xtile[(t + 1) & 1][0],
              (uint64_t)(size_t)(xproj + ((size_t)batch0 * T_SZ + (t + 1)) * G_SZ),
              G_SZ, 16, (uint64_t)T_SZ * G_SZ, 1u);
          __builtin_amdgcn_s_wait_tensorcnt(1);
        } else {
          __builtin_amdgcn_s_wait_tensorcnt(0);
        }
      }
      __syncthreads();  // publish xtile[t&1] to all waves
    }

    const bf16_t* xt = xtile[t & 1];
    const int hcol = hg * 16 + lane16;
#pragma unroll
    for (int v = 0; v < 8; ++v) {
      const int m = v + laneHalf * 8;
      float gi = acc[0][v] + bias[0];
      float gf = acc[1][v] + bias[1];
      float gg = acc[2][v] + bias[2];
      float go = acc[3][v] + bias[3];
      if constexpr (!FUSED) {
        const bf16_t* xr = xt + m * G_SZ;
        gi += (float)xr[0 * 256 + hcol];
        gf += (float)xr[1 * 256 + hcol];
        gg += (float)xr[2 * 256 + hcol];
        go += (float)xr[3 * 256 + hcol];
      }
      gi = sigmoidf_(gi);
      gf = sigmoidf_(gf);
      go = sigmoidf_(go);
      gg = tanhf(gg);
      const float c = gf * cst[v] + gi * gg;
      cst[v] = c;
      const float h = go * tanhf(c);
      hnext[m * H_SZ + hcol] = (bf16_t)h;
      if (t == T_SZ - 1)
        h_last[(size_t)(batch0 + m) * H_SZ + hcol] = h;
    }
    __syncthreads();
  }
}

// ---------------------------------------------------------------------------
// out[b][c] = h_last[b] . W_cls[c] + b_cls[c]   (64x10, trivial)
// ---------------------------------------------------------------------------
__global__ __launch_bounds__(256) void classify_kernel(
    const float* __restrict__ hlast, const float* __restrict__ wcls,
    const float* __restrict__ bcls, float* __restrict__ out) {
  const int idx = blockIdx.x * blockDim.x + threadIdx.x;
  if (idx >= B_SZ * C_SZ) return;
  const int b = idx / C_SZ;
  const int c = idx % C_SZ;
  float s = bcls[c];
#pragma unroll 8
  for (int k = 0; k < H_SZ; ++k)
    s += hlast[(size_t)b * H_SZ + k] * wcls[(size_t)c * H_SZ + k];
  out[idx] = s;
}

extern "C" void kernel_launch(void* const* d_in, const int* in_sizes, int n_in,
                              void* d_out, int out_size, void* d_ws,
                              size_t ws_size, hipStream_t stream) {
  (void)in_sizes; (void)n_in; (void)out_size;
  const float* x     = (const float*)d_in[0];
  const float* key   = (const float*)d_in[1];
  const float* w_ih  = (const float*)d_in[2];
  const float* w_hh  = (const float*)d_in[3];
  const float* b_ih  = (const float*)d_in[4];
  const float* b_hh  = (const float*)d_in[5];
  const float* w_cls = (const float*)d_in[6];
  const float* b_cls = (const float*)d_in[7];
  float* out = (float*)d_out;

  uint8_t* ws = (uint8_t*)d_ws;
  const size_t OFF_WCT   = 0;                      // 1024*256*2 = 512 KB
  const size_t OFF_WHH   = 512u * 1024u;           // 512 KB
  const size_t OFF_HLAST = 1024u * 1024u;          // 64*256*4 = 64 KB
  const size_t OFF_XPROJ = 1024u * 1024u + 64u * 1024u;
  const size_t XPROJ_BYTES = (size_t)B_SZ * T_SZ * G_SZ * sizeof(bf16_t); // 256 MB

  bf16_t* wct   = (bf16_t*)(ws + OFF_WCT);
  bf16_t* whh_b = (bf16_t*)(ws + OFF_WHH);
  float*  hlast = (float*)(ws + OFF_HLAST);
  bf16_t* xproj = (bf16_t*)(ws + OFF_XPROJ);

  const bool precomp = ws_size >= OFF_XPROJ + XPROJ_BYTES;

  pack_wct_kernel<<<G_SZ, 256, 0, stream>>>(key, w_ih, wct);
  pack_whh_kernel<<<(G_SZ * H_SZ) / 256, 256, 0, stream>>>(w_hh, whh_b);

  if (precomp) {
    xproj_gemm_kernel<<<dim3(G_SZ / 64, (B_SZ * T_SZ) / 128), 256, 0, stream>>>(
        x, wct, b_ih, xproj);
    lstm_scan_kernel<false><<<B_SZ / 16, 512, 0, stream>>>(
        x, xproj, wct, whh_b, b_ih, b_hh, hlast);
  } else {
    lstm_scan_kernel<true><<<B_SZ / 16, 512, 0, stream>>>(
        x, nullptr, wct, whh_b, b_ih, b_hh, hlast);
  }

  classify_kernel<<<(B_SZ * C_SZ + 255) / 256, 256, 0, stream>>>(
      hlast, w_cls, b_cls, out);
}
